// LSM_partition_25881472926530
// MI455X (gfx1250) — matrix operand order repaired and checked by reading, not compiled, not run on hardware
//
#include <hip/hip_runtime.h>

typedef __bf16 bf16_t;
typedef __attribute__((ext_vector_type(8)))  __bf16 v8bf;
typedef __attribute__((ext_vector_type(16))) __bf16 v16bf;
typedef __attribute__((ext_vector_type(8)))  float  v8f;

#define T_STEPS 192
#define BATCH   128
#define IN_DIM  512
#define RDIM    1024
#define OUT_DIM 100
#define PS      64
#define LSM_ALPHA 0.9f
#define LSM_BETA  0.9f
#define LSM_THR   1.0f

#define NBLOCKS 64           // persistent grid: 64 blocks x 8 waves = 512 tile-waves
#define LDA_X (IN_DIM + 8)   // LDS padding: +8 bf16 (4 dwords) rotates banks per row
#define LDA_S (RDIM + 8)

// ---------------- WMMA fragment loaders (bf16, 16x16x32) ----------------
// A-matrix 16x32 (MxK), row-major source (global or LDS) with leading dim lda:
//   lanes 0-15 : row M=lane,     elems 0..7 = K 0..7,  elems 8..15 = K 16..23
//   lanes16-31 : row M=lane-16,  elems 0..7 = K 8..15, elems 8..15 = K 24..31
__device__ __forceinline__ v16bf load_a_frag(const bf16_t* __restrict__ A, int lda) {
  int lane = threadIdx.x & 31;
  int half = lane >> 4;
  int row  = lane & 15;
  const bf16_t* p = A + (size_t)row * lda + half * 8;
  v8bf lo = *(const v8bf*)(p);       // K = k0 .. k0+7
  v8bf hi = *(const v8bf*)(p + 16);  // K = k0+16 .. k0+23
  return __builtin_shufflevector(lo, hi, 0,1,2,3,4,5,6,7,8,9,10,11,12,13,14,15);
}

// B-matrix 32x16 (KxN) with B[k][n] = W[n][k] (computing spk @ W^T):
//   per lane: col n = lane&15, elems 0..7 = K (lane>=16)*8 + 0..7, elems 8..15 = K 16 + (lane>=16)*8 + 0..7.
__device__ __forceinline__ v16bf load_b_frag(const bf16_t* __restrict__ W, int ldw) {
  int lane = threadIdx.x & 31;
  int half = lane >> 4;
  int col  = lane & 15;
  const bf16_t* p = W + (size_t)col * ldw + half * 8;
  v8bf lo = *(const v8bf*)(p);
  v8bf hi = *(const v8bf*)(p + 16);
  return __builtin_shufflevector(lo, hi, 0,1,2,3,4,5,6,7,8,9,10,11,12,13,14,15);
}

__device__ __forceinline__ v8f gemm_acc(v8f c, const bf16_t* __restrict__ A, int lda,
                                        const bf16_t* __restrict__ W, int ldw, int K) {
  #pragma unroll 4
  for (int k = 0; k < K; k += 32) {
    v16bf a = load_a_frag(A + k, lda);
    v16bf b = load_b_frag(W + k, ldw);
    c = __builtin_amdgcn_wmma_f32_16x16x32_bf16(false, a, false, b, (short)0, c,
                                                false, false);
  }
  return c;
}

// Cooperative 256-thread copy of a 16-row tile (row-major) into padded LDS.
__device__ __forceinline__ void stage_tile(const bf16_t* __restrict__ g, int ldg,
                                           bf16_t* __restrict__ l, int ldl, int cols) {
  int cpr = cols >> 3;                 // v8bf chunks per row
  int total = 16 * cpr;
  for (int c = threadIdx.x; c < total; c += 256) {
    int row = c / cpr;
    int cc  = c - row * cpr;
    *(v8bf*)(l + row * ldl + cc * 8) = *(const v8bf*)(g + (size_t)row * ldg + cc * 8);
  }
}

// Device-wide barrier for a co-resident persistent grid (release/acquire via L2).
__device__ __forceinline__ void grid_sync(unsigned* cnt, unsigned target) {
  __threadfence();
  __syncthreads();
  if (threadIdx.x == 0) {
    __hip_atomic_fetch_add(cnt, 1u, __ATOMIC_RELEASE, __HIP_MEMORY_SCOPE_AGENT);
    while (__hip_atomic_load(cnt, __ATOMIC_ACQUIRE, __HIP_MEMORY_SCOPE_AGENT) < target) {
      __builtin_amdgcn_s_sleep(2);
    }
  }
  __syncthreads();
}

// ---------------- kernels ----------------
__global__ void cvt_f32_bf16(const float* __restrict__ in, bf16_t* __restrict__ out, int n) {
  int i = blockIdx.x * blockDim.x + threadIdx.x;
  if (i < n) out[i] = (bf16_t)in[i];
}

__global__ void zero_u32(unsigned* p) { *p = 0u; }

// Persistent fused scan: all 192 steps in one kernel; syn/mem live in VGPRs.
// Wave tile: 16x16 (batch x reservoir). Block stages shared A-tiles in LDS.
__global__ __launch_bounds__(256) void lsm_scan(
    const bf16_t* __restrict__ x_bf,   // [T,B,IN]
    const bf16_t* __restrict__ Win,    // [3,R,IN]
    const bf16_t* __restrict__ Wrec,   // [R,R]
    const bf16_t* __restrict__ Wlin,   // [R,R]
    const float*  __restrict__ b_in,   // [R]
    const float*  __restrict__ b_rec,  // [R]
    bf16_t* __restrict__ spk_bf,       // [T,B,R] bf16 spike history
    float*  __restrict__ spk_out,      // [T,B,R] fp32 output
    unsigned* __restrict__ counter) {
  __shared__ bf16_t lX [16 * LDA_X];   // x[t] tile        (16.6 KB)
  __shared__ bf16_t lS1[16 * LDA_S];   // spk_{t-1} tile   (33.0 KB)

  const int wave = threadIdx.x >> 5;
  const int bt   = (blockIdx.x >> 3) << 4;                 // shared by all 8 waves
  const int rt   = (((blockIdx.x & 7) << 3) + wave) << 4;  // per-wave reservoir tile

  const int lane = threadIdx.x & 31;
  const int half = lane >> 4;
  const int r    = rt + (lane & 15);
  const float bias = b_in[r] + b_rec[r];

  float syn[8], mem[8];
  #pragma unroll
  for (int i = 0; i < 8; ++i) { syn[i] = 0.0f; mem[i] = 0.0f; }

  for (int t = 0; t < T_STEPS; ++t) {
    const int p = t / PS;

    // ---- stage shared A-operands into LDS (8x reuse within block) ----
    stage_tile(x_bf + ((size_t)t * BATCH + bt) * IN_DIM, IN_DIM, lX, LDA_X, IN_DIM);
    if (t > 0)
      stage_tile(spk_bf + ((size_t)(t - 1) * BATCH + bt) * RDIM, RDIM, lS1, LDA_S, RDIM);
    __syncthreads();

    // ---- fused GEMMs: cur + recurrent + delayed feedback ----
    v8f c = {};
    c = gemm_acc(c, lX, LDA_X, Win + ((size_t)p * RDIM + rt) * IN_DIM, IN_DIM, IN_DIM);
    if (t > 0)
      c = gemm_acc(c, lS1, LDA_S, Wrec + (size_t)rt * RDIM, RDIM, RDIM);
    if (t > PS)  // strict t > ps, as in reference
      c = gemm_acc(c, spk_bf + ((size_t)(t - PS) * BATCH + bt) * RDIM, RDIM,
                   Wlin + (size_t)rt * RDIM, RDIM, RDIM);

    // ---- RSynaptic neuron update (state in registers) ----
    #pragma unroll
    for (int i = 0; i < 8; ++i) {
      int b = bt + half * 8 + i;                 // D layout: M = i + 8*(lane>=16)
      float s = syn[i];
      float m = mem[i];
      float reset = (m > LSM_THR) ? 1.0f : 0.0f; // reset from pre-update mem
      s = LSM_ALPHA * s + c[i] + bias;
      m = LSM_BETA * m + s - reset * LSM_THR;
      float spk = (m > LSM_THR) ? 1.0f : 0.0f;   // Heaviside(mem - thr)
      syn[i] = s;
      mem[i] = m;
      size_t tidx = ((size_t)t * BATCH + b) * RDIM + r;
      spk_out[tidx] = spk;
      spk_bf[tidx]  = (bf16_t)spk;               // exact in bf16
    }

    // ---- device-wide step barrier ----
    grid_sync(counter, (unsigned)NBLOCKS * (unsigned)(t + 1));
  }
}

// ro_in[b, p*R + r] = mean over ps steps of spk_rec[p*ps + s, b, r]
__global__ void rates_kernel(const float* __restrict__ spk, float* __restrict__ ro) {
  int i = blockIdx.x * blockDim.x + threadIdx.x;
  const int NPR = 3 * RDIM;
  if (i >= BATCH * NPR) return;
  int b  = i / NPR;
  int pr = i - b * NPR;
  int p  = pr / RDIM;
  int r  = pr - p * RDIM;
  float s = 0.0f;
  #pragma unroll 4
  for (int k = 0; k < PS; ++k)
    s += spk[((size_t)(p * PS + k) * BATCH + b) * RDIM + r];
  ro[i] = s * (1.0f / PS);
}

// out[b,:] = softmax(ro[b,:] @ W_ro^T + b_ro). One block per batch row.
__global__ __launch_bounds__(128) void readout_kernel(
    const float* __restrict__ ro, const float* __restrict__ Wro,
    const float* __restrict__ bro, float* __restrict__ out) {
  __shared__ float red[128];
  int b = blockIdx.x, o = threadIdx.x;
  const int K = 3 * RDIM;
  float l = -3.402823466e38f;
  if (o < OUT_DIM) {
    l = bro[o];
    const float* row = ro + (size_t)b * K;
    const float* wr  = Wro + (size_t)o * K;
    #pragma unroll 4
    for (int k = 0; k < K; ++k) l += row[k] * wr[k];
  }
  red[o] = l;
  __syncthreads();
  for (int s = 64; s > 0; s >>= 1) {
    if (o < s) red[o] = fmaxf(red[o], red[o + s]);
    __syncthreads();
  }
  float mx = red[0];
  __syncthreads();
  float e = (o < OUT_DIM) ? __expf(l - mx) : 0.0f;
  red[o] = e;
  __syncthreads();
  for (int s = 64; s > 0; s >>= 1) {
    if (o < s) red[o] += red[o + s];
    __syncthreads();
  }
  if (o < OUT_DIM) out[(size_t)b * OUT_DIM + o] = e / red[0];
}

// ---------------- host launcher ----------------
extern "C" void kernel_launch(void* const* d_in, const int* in_sizes, int n_in,
                              void* d_out, int out_size, void* d_ws, size_t ws_size,
                              hipStream_t stream) {
  (void)in_sizes; (void)n_in; (void)out_size; (void)ws_size;
  const float* x     = (const float*)d_in[0];  // [T,B,IN]
  const float* W_in  = (const float*)d_in[1];  // [3,R,IN]
  const float* b_in  = (const float*)d_in[2];  // [R]
  const float* W_lin = (const float*)d_in[3];  // [R,R]
  const float* W_rec = (const float*)d_in[4];  // [R,R]
  const float* b_rec = (const float*)d_in[5];  // [R]
  const float* W_ro  = (const float*)d_in[6];  // [OUT, 3R]
  const float* b_ro  = (const float*)d_in[7];  // [OUT]

  float* out     = (float*)d_out;              // [B,OUT]
  float* spk_rec = out + BATCH * OUT_DIM;      // [T,B,R]

  // workspace carve-up
  char* ws = (char*)d_ws;
  size_t off = 0;
  auto carve = [&](size_t bytes) -> char* {
    char* p = ws + off;
    off = (off + bytes + 255) & ~(size_t)255;
    return p;
  };
  const size_t nX   = (size_t)T_STEPS * BATCH * IN_DIM;
  const size_t nWin = (size_t)3 * RDIM * IN_DIM;
  const size_t nWsq = (size_t)RDIM * RDIM;
  const size_t nSpk = (size_t)T_STEPS * BATCH * RDIM;

  bf16_t*   x_bf    = (bf16_t*)carve(nX * sizeof(bf16_t));
  bf16_t*   Win_bf  = (bf16_t*)carve(nWin * sizeof(bf16_t));
  bf16_t*   Wrec_bf = (bf16_t*)carve(nWsq * sizeof(bf16_t));
  bf16_t*   Wlin_bf = (bf16_t*)carve(nWsq * sizeof(bf16_t));
  bf16_t*   spk_bf  = (bf16_t*)carve(nSpk * sizeof(bf16_t));
  float*    ro_in   = (float*)carve((size_t)BATCH * 3 * RDIM * sizeof(float));
  unsigned* counter = (unsigned*)carve(256);

  // 1) precision conversion (bf16 unlocks v_wmma_f32_16x16x32_bf16; spikes exact)
  cvt_f32_bf16<<<(int)((nX   + 255) / 256), 256, 0, stream>>>(x,     x_bf,    (int)nX);
  cvt_f32_bf16<<<(int)((nWin + 255) / 256), 256, 0, stream>>>(W_in,  Win_bf,  (int)nWin);
  cvt_f32_bf16<<<(int)((nWsq + 255) / 256), 256, 0, stream>>>(W_rec, Wrec_bf, (int)nWsq);
  cvt_f32_bf16<<<(int)((nWsq + 255) / 256), 256, 0, stream>>>(W_lin, Wlin_bf, (int)nWsq);
  zero_u32<<<1, 1, 0, stream>>>(counter);

  // 2) persistent fused scan: one launch, 192 internally-synchronized steps
  lsm_scan<<<NBLOCKS, 256, 0, stream>>>(x_bf, Win_bf, Wrec_bf, Wlin_bf, b_in, b_rec,
                                        spk_bf, spk_rec, counter);

  // 3) partition spike-rate means + softmax readout
  rates_kernel<<<(BATCH * 3 * RDIM + 255) / 256, 256, 0, stream>>>(spk_rec, ro_in);
  readout_kernel<<<BATCH, 128, 0, stream>>>(ro_in, W_ro, b_ro, out);
}
